// GatedLinearCrossAttention_35373350650143
// MI455X (gfx1250) — compile-verified, matrix operand order checked by reading, not run on hardware
//
#include <hip/hip_runtime.h>
#include <hip/hip_bf16.h>

// ---------------------------------------------------------------------------
// Gated Linear Cross Attention for MI455X (gfx1250), bf16 WMMA pipeline.
// GEMM A-tiles stream through the Tensor Data Mover (tensor_load_to_lds,
// TENSORcnt, LDS pitch applied in-flight via D# padding); B-tiles go through
// a register pipeline into a transposed LDS layout so fragments stay
// ds_load_b128. Double-buffered LDS, one barrier per K-slice.
// B=4, LQ=LK=2048, D=1024, H=16, QK=512, M=32, HD=64, R=16
// ---------------------------------------------------------------------------

#define B_    4
#define LQ_   2048
#define LK_   2048
#define D_    1024
#define H_    16
#define QK_   512
#define MD_   32          // qk head dim
#define HD_   64          // v head dim
#define R_    16
#define ROWS_ (B_ * LQ_)  // 8192 flattened rows
#define GLN_  16.0f
#define RMSEPS_ 1e-6f

typedef __attribute__((ext_vector_type(16))) __bf16 v16bf;
typedef __attribute__((ext_vector_type(8)))  float  v8f;
typedef __attribute__((ext_vector_type(4)))  unsigned int u32x4;
typedef __attribute__((ext_vector_type(8)))  int          i32x8;
typedef __attribute__((ext_vector_type(4)))  int          i32x4;

union FragBF {
  v16bf v;
  unsigned int   u[8];
  unsigned short h[16];
};

// K index of the low bf16 of 32-bit pair i (0..7) for a lane half, per the
// CDNA5 16-bit A-matrix 16x32 layout (lanes 0-15: K 0-7/16-23, lanes 16-31:
// K 8-15/24-31, two elements per VGPR). B fragments use the mirrored layout
// (lane = column N, pairs over K).
__device__ __forceinline__ int pairK(int i, int half) {
  return ((i & 4) << 2) + (half << 3) + ((i & 3) << 1);
}

// ---------------------------------------------------------------------------
// Tensor DMA descriptor helpers (ISA 08_async_tensor §8.3/§8.4).
// Group0: count=1, lds_addr[63:32], global_addr[120:64], type=2 [127:126].
// Group1: data_size=2B, pad_enable, pad_interval/amount, tensor/tile dims,
//         dim0 stride (fields straddle dword boundaries).
// ---------------------------------------------------------------------------
__device__ __forceinline__ u32x4 tdm_g0(unsigned lds_addr, unsigned long long ga) {
  u32x4 g;
  g[0] = 1u;                                       // count=1, user mode
  g[1] = lds_addr;                                 // LDS byte address
  g[2] = (unsigned)ga;                             // global addr [31:0]
  g[3] = (unsigned)((ga >> 32) & 0x01FFFFFFu) | (2u << 30);  // [56:32] | type=2
  return g;
}

__device__ __forceinline__ i32x8 tdm_g1(unsigned ctrl, unsigned td0, unsigned td1,
                                        unsigned tile0, unsigned tile1,
                                        unsigned long long stride0) {
  i32x8 g;
  g[0] = (int)ctrl;                                // mask=0|data_size|pad cfg
  g[1] = (int)((td0 & 0xFFFFu) << 16);             // abar=0 | tensor_dim0 lo
  g[2] = (int)((td0 >> 16) | ((td1 & 0xFFFFu) << 16));
  g[3] = (int)((td1 >> 16) | (tile0 << 16));       // tensor_dim1 hi | tile_dim0
  g[4] = (int)(tile1 & 0xFFFFu);                   // tile_dim1 | tile_dim2=0
  g[5] = (int)(unsigned)(stride0 & 0xFFFFFFFFu);   // dim0 stride lo
  g[6] = (int)(unsigned)((stride0 >> 32) & 0xFFFFu); // stride hi | stride1 lo=0
  g[7] = 0;
  return g;
}

// ctrl word: data_size=1 (2 bytes) at [17:16], pad_enable at [20],
// pad_interval at [24:22] (value p => pad every 2^(p+1) dwords),
// pad_amount at [31:25] (value a => a+1 dwords of pad).
// A tile rows are 32 bf16 = 16 dwords; 4 dwords pad -> LDS pitch 40 bf16.
#define TDM_CTRL_A ((1u << 16) | (1u << 20) | (3u << 22) | (3u << 25))

// ---------------------------------------------------------------------------
// fp32 -> bf16 conversion
// ---------------------------------------------------------------------------
__global__ void cvt_f32_bf16(const float* __restrict__ src,
                             __bf16* __restrict__ dst, int n) {
  int i = blockIdx.x * blockDim.x + threadIdx.x;
  int stride = gridDim.x * blockDim.x;
  for (; i < n; i += stride) dst[i] = (__bf16)src[i];
}

// ---------------------------------------------------------------------------
// Generic bf16 GEMM: C[M,N] = A[M,K] @ B[K,N]; fp32 accumulate.
// Workgroup tile 128x128, BK=32 (one v_wmma_f32_16x16x32_bf16 K-step).
// 8 waves, each computes a 32x64 sub-tile (2x4 WMMA tiles).
// Requires M%128==0, N%128==0, K%32==0 (true for all uses here).
// ---------------------------------------------------------------------------
#define BM 128
#define BN 128
#define BK 32
#define LP 40    // LDS pitch in bf16 (80B rows, conflict-free for 16 lanes)

__global__ __launch_bounds__(256)
void gemm_bf16_kernel(const __bf16* __restrict__ A, const __bf16* __restrict__ Bm,
                      void* __restrict__ C, int M, int N, int K, int outBf16) {
  const int tid   = threadIdx.x;
  const int lane  = tid & 31;
  const int wid   = tid >> 5;
  const int half  = lane >> 4;
  const int q15   = lane & 15;
  const int waveM = wid >> 1;            // 0..3
  const int waveN = wid & 1;             // 0..1
  const int tileM = blockIdx.y * BM;
  const int tileN = blockIdx.x * BN;

  // B-tile cooperative slots (512 slots of 8 bf16 / 256 threads = 2 each).
  const int bK0 = tid >> 4,         bNo0 = (tid & 15) << 3;
  const int bK1 = (tid + 256) >> 4, bNo1 = (tid & 15) << 3;

  v8f acc[2][4];
  for (int mt = 0; mt < 2; ++mt)
    for (int nt = 0; nt < 4; ++nt)
      acc[mt][nt] = (v8f)(0.0f);

#if defined(__gfx1250__) && __has_builtin(__builtin_amdgcn_tensor_load_to_lds) && \
    __has_builtin(__builtin_amdgcn_s_wait_tensorcnt)
  // ============ A via TDM, B via register pipeline; double-buffered ============
  __shared__ __align__(16) __bf16 Al[2][BM * LP];  // [row][k], pitch 40 (TDM pad)
  __shared__ __align__(16) __bf16 Bt[2][BN * LP];  // [col][k], pitch 40 (transposed)

  const unsigned ldsA[2] = { (unsigned)(size_t)(void*)&Al[0][0],
                             (unsigned)(size_t)(void*)&Al[1][0] };
  const i32x4 gz4 = { 0, 0, 0, 0 };
  const i32x8 gz8 = { 0, 0, 0, 0, 0, 0, 0, 0 };
  const i32x8 g1A = tdm_g1(TDM_CTRL_A, (unsigned)K, (unsigned)M, BK, BM,
                           (unsigned long long)K);

  // ---- prologue: slice 0 ----
  if (wid == 0) {
    unsigned long long gaA = (unsigned long long)(size_t)(A + (size_t)tileM * K);
    __builtin_amdgcn_tensor_load_to_lds(tdm_g0(ldsA[0], gaA), g1A, gz4, gz4, gz8, 0);
  }
  {
    union { uint4 v; __bf16 b[8]; } d0, d1;
    d0.v = *(const uint4*)(Bm + (size_t)bK0 * N + tileN + bNo0);
    d1.v = *(const uint4*)(Bm + (size_t)bK1 * N + tileN + bNo1);
    for (int j = 0; j < 8; ++j) Bt[0][(bNo0 + j) * LP + bK0] = d0.b[j];
    for (int j = 0; j < 8; ++j) Bt[0][(bNo1 + j) * LP + bK1] = d1.b[j];
  }
  if (wid == 0) __builtin_amdgcn_s_wait_tensorcnt(0);
  __syncthreads();

  for (int kt = 0; kt < K; kt += BK) {
    const int cur = (kt >> 5) & 1;
    const int nxt = cur ^ 1;
    const bool more = (kt + BK) < K;

    // Kick off next slice: async TDM for A, global loads into regs for B.
    uint4 rb0, rb1;
    if (more) {
      if (wid == 0) {
        unsigned long long gaA =
            (unsigned long long)(size_t)(A + (size_t)tileM * K + kt + BK);
        __builtin_amdgcn_tensor_load_to_lds(tdm_g0(ldsA[nxt], gaA), g1A,
                                            gz4, gz4, gz8, 0);
      }
      rb0 = *(const uint4*)(Bm + (size_t)(kt + BK + bK0) * N + tileN + bNo0);
      rb1 = *(const uint4*)(Bm + (size_t)(kt + BK + bK1) * N + tileN + bNo1);
    }

    // ---- compute slice `cur` (hides TDM + global-load latency) ----
    FragBF af[2], bf[4];
    for (int mt = 0; mt < 2; ++mt) {
      int m = waveM * 32 + mt * 16 + q15;
      for (int i = 0; i < 8; ++i)
        af[mt].u[i] = *(const unsigned int*)(&Al[cur][0] + m * LP + pairK(i, half));
    }
    for (int nt = 0; nt < 4; ++nt) {
      int n = waveN * 64 + nt * 16 + q15;
      for (int i = 0; i < 8; ++i)
        bf[nt].u[i] = *(const unsigned int*)(&Bt[cur][0] + n * LP + pairK(i, half));
    }
    for (int mt = 0; mt < 2; ++mt)
      for (int nt = 0; nt < 4; ++nt)
        acc[mt][nt] = __builtin_amdgcn_wmma_f32_16x16x32_bf16(
            false, af[mt].v, false, bf[nt].v, (short)0, acc[mt][nt],
            false, false);

    // Stage next B slice (transposed) into the other buffer; nobody reads
    // `nxt` until after the barrier below.
    if (more) {
      union { uint4 v; __bf16 b[8]; } d0, d1;
      d0.v = rb0; d1.v = rb1;
      for (int j = 0; j < 8; ++j) Bt[nxt][(bNo0 + j) * LP + bK0] = d0.b[j];
      for (int j = 0; j < 8; ++j) Bt[nxt][(bNo1 + j) * LP + bK1] = d1.b[j];
      if (wid == 0) __builtin_amdgcn_s_wait_tensorcnt(0);  // A[nxt] landed
    }
    __syncthreads();   // hand off: everyone done with cur, nxt fully staged
  }
#else
  // ================== fallback: register-staged pipeline ==================
  __shared__ __align__(16) __bf16 Al[BM * LP];     // [row][k]
  __shared__ __align__(16) __bf16 Bt[BN * LP];     // [col][k] (transposed)

  const int aRow0 = tid >> 2,         aKo0 = (tid & 3) << 3;
  const int aRow1 = (tid + 256) >> 2, aKo1 = (tid & 3) << 3;

  uint4 ra0 = *(const uint4*)(A + (size_t)(tileM + aRow0) * K + aKo0);
  uint4 ra1 = *(const uint4*)(A + (size_t)(tileM + aRow1) * K + aKo1);
  uint4 rb0 = *(const uint4*)(Bm + (size_t)bK0 * N + tileN + bNo0);
  uint4 rb1 = *(const uint4*)(Bm + (size_t)bK1 * N + tileN + bNo1);

  for (int kt = 0; kt < K; kt += BK) {
    *(uint4*)(Al + aRow0 * LP + aKo0) = ra0;
    *(uint4*)(Al + aRow1 * LP + aKo1) = ra1;
    {
      union { uint4 v; __bf16 b[8]; } d0, d1;
      d0.v = rb0; d1.v = rb1;
      for (int j = 0; j < 8; ++j) Bt[(bNo0 + j) * LP + bK0] = d0.b[j];
      for (int j = 0; j < 8; ++j) Bt[(bNo1 + j) * LP + bK1] = d1.b[j];
    }
    __syncthreads();

    const int kn = kt + BK;
    if (kn < K) {
      ra0 = *(const uint4*)(A + (size_t)(tileM + aRow0) * K + kn + aKo0);
      ra1 = *(const uint4*)(A + (size_t)(tileM + aRow1) * K + kn + aKo1);
      rb0 = *(const uint4*)(Bm + (size_t)(kn + bK0) * N + tileN + bNo0);
      rb1 = *(const uint4*)(Bm + (size_t)(kn + bK1) * N + tileN + bNo1);
    }

    FragBF af[2], bf[4];
    for (int mt = 0; mt < 2; ++mt) {
      int m = waveM * 32 + mt * 16 + q15;
      for (int i = 0; i < 8; ++i)
        af[mt].u[i] = *(const unsigned int*)(Al + m * LP + pairK(i, half));
    }
    for (int nt = 0; nt < 4; ++nt) {
      int n = waveN * 64 + nt * 16 + q15;
      for (int i = 0; i < 8; ++i)
        bf[nt].u[i] = *(const unsigned int*)(Bt + n * LP + pairK(i, half));
    }
    for (int mt = 0; mt < 2; ++mt)
      for (int nt = 0; nt < 4; ++nt)
        acc[mt][nt] = __builtin_amdgcn_wmma_f32_16x16x32_bf16(
            false, af[mt].v, false, bf[nt].v, (short)0, acc[mt][nt],
            false, false);
    __syncthreads();
  }
#endif

  // Epilogue. C/D layout: VGPR g holds (M = g + 8*half, N = q15) per tile.
  for (int mt = 0; mt < 2; ++mt) {
    for (int nt = 0; nt < 4; ++nt) {
      int col = tileN + waveN * 64 + nt * 16 + q15;
      for (int g = 0; g < 8; ++g) {
        int row = tileM + waveM * 32 + mt * 16 + g + 8 * half;
        float v = acc[mt][nt][g];
        if (outBf16) ((__bf16*)C)[(size_t)row * N + col] = (__bf16)v;
        else         ((float*)C)[(size_t)row * N + col]  = v;
      }
    }
  }
}

// ---------------------------------------------------------------------------
// gk = logsigmoid((value @ Wgk1) @ Wgk2 + bgk) / 16   (R=16 low rank, VALU)
// One block per flattened row l (8192 blocks).
// ---------------------------------------------------------------------------
__global__ __launch_bounds__(256)
void gk_proj_kernel(const float* __restrict__ value,
                    const float* __restrict__ Wgk1,
                    const float* __restrict__ Wgk2,
                    const float* __restrict__ bgk,
                    float* __restrict__ gk) {
  __shared__ float part[256];
  __shared__ float t[R_];
  const int l   = blockIdx.x;
  const int tid = threadIdx.x;
  const int i0  = tid & 15;
  const int j   = tid >> 4;

  const float* vr = value + (size_t)l * D_;
  float p = 0.f;
  for (int i = i0; i < D_; i += 16) p += vr[i] * Wgk1[i * R_ + j];
  part[tid] = p;
  __syncthreads();
  if (tid < R_) {
    float s = 0.f;
    for (int i = 0; i < 16; ++i) s += part[tid * 16 + i];
    t[tid] = s;
  }
  __syncthreads();

  for (int n = tid; n < QK_; n += 256) {
    float acc = bgk[n];
    for (int r = 0; r < R_; ++r) acc += t[r] * Wgk2[r * QK_ + n];
    // logsigmoid(x) = min(x,0) - log1p(exp(-|x|))
    float ls = fminf(acc, 0.f) - log1pf(expf(-fabsf(acc)));
    gk[(size_t)l * QK_ + n] = ls * (1.0f / GLN_);
  }
}

// ---------------------------------------------------------------------------
// decay = exp(flip(cumsum(gk, L), L)); kd = k * decay  (three-phase scan)
// Columns c = h*32+m (512 per batch); chunked over L (16 chunks x 128 rows).
// ---------------------------------------------------------------------------
#define NCH 16
#define CHL (LK_ / NCH)   // 128

__global__ __launch_bounds__(512)
void scan_chunksum(const float* __restrict__ gk, float* __restrict__ cs) {
  const int c = threadIdx.x, ch = blockIdx.x, b = blockIdx.y;
  float s = 0.f;
  for (int r = 0; r < CHL; ++r)
    s += gk[(size_t)(b * LK_ + ch * CHL + r) * QK_ + c];
  cs[(size_t)(b * NCH + ch) * QK_ + c] = s;
}

__global__ __launch_bounds__(512)
void scan_chunkoff(const float* __restrict__ cs, float* __restrict__ co) {
  const int c = threadIdx.x, b = blockIdx.x;
  float off = 0.f;
  for (int ch = 0; ch < NCH; ++ch) {
    co[(size_t)(b * NCH + ch) * QK_ + c] = off;
    off += cs[(size_t)(b * NCH + ch) * QK_ + c];
  }
}

__global__ __launch_bounds__(512)
void scan_apply(const float* __restrict__ gk, const float* __restrict__ co,
                const __bf16* __restrict__ Kp, __bf16* __restrict__ kd) {
  const int c = threadIdx.x, ch = blockIdx.x, b = blockIdx.y;
  float run = co[(size_t)(b * NCH + ch) * QK_ + c];
  for (int r = 0; r < CHL; ++r) {
    int p = ch * CHL + r;
    run += gk[(size_t)(b * LK_ + p) * QK_ + c];     // inclusive cumsum c[p]
    int lq = LK_ - 1 - p;                            // decay[lq] = exp(c[p])
    size_t idx = (size_t)(b * LK_ + lq) * QK_ + c;
    kd[idx] = (__bf16)((float)Kp[idx] * expf(run));
  }
}

// ---------------------------------------------------------------------------
// State: s[b,h] (32x64) = kd[b,:,h]^T @ v[b,:,h]  (K = LK = 2048)
// 4 waves split K, LDS tree reduce.
// ---------------------------------------------------------------------------
__global__ __launch_bounds__(128)
void state_kernel(const __bf16* __restrict__ kd, const __bf16* __restrict__ Vp,
                  __bf16* __restrict__ sbf) {
  __shared__ float red[4][MD_ * HD_];
  const int h = blockIdx.x, b = blockIdx.y;
  const int tid  = threadIdx.x;
  const int wid  = tid >> 5;
  const int lane = tid & 31;
  const int half = lane >> 4;
  const int q15  = lane & 15;

  v8f acc[2][4];
  for (int mt = 0; mt < 2; ++mt)
    for (int nt = 0; nt < 4; ++nt) acc[mt][nt] = (v8f)(0.0f);

  const size_t kdB = (size_t)b * LK_ * QK_;
  const size_t vB  = (size_t)b * LK_ * D_;

  for (int kt = wid * (LK_ / 4); kt < (wid + 1) * (LK_ / 4); kt += 32) {
    FragBF af[2], bf[4];
    for (int mt = 0; mt < 2; ++mt) {            // A = kd^T: A[m,l]=kd[l, h*32+m]
      int c = h * MD_ + mt * 16 + q15;
      for (int i = 0; i < 8; ++i) {
        int l = kt + pairK(i, half);
        af[mt].h[2 * i]     = *(const unsigned short*)(kd + kdB + (size_t)l * QK_ + c);
        af[mt].h[2 * i + 1] = *(const unsigned short*)(kd + kdB + (size_t)(l + 1) * QK_ + c);
      }
    }
    for (int nt = 0; nt < 4; ++nt) {            // B = v: B[l,n]=Vp[l, h*64+n]
      int c = h * HD_ + nt * 16 + q15;
      for (int i = 0; i < 8; ++i) {
        int l = kt + pairK(i, half);
        bf[nt].h[2 * i]     = *(const unsigned short*)(Vp + vB + (size_t)l * D_ + c);
        bf[nt].h[2 * i + 1] = *(const unsigned short*)(Vp + vB + (size_t)(l + 1) * D_ + c);
      }
    }
    for (int mt = 0; mt < 2; ++mt)
      for (int nt = 0; nt < 4; ++nt)
        acc[mt][nt] = __builtin_amdgcn_wmma_f32_16x16x32_bf16(
            false, af[mt].v, false, bf[nt].v, (short)0, acc[mt][nt],
            false, false);
  }

  for (int mt = 0; mt < 2; ++mt)
    for (int nt = 0; nt < 4; ++nt)
      for (int g = 0; g < 8; ++g) {
        int m = mt * 16 + g + 8 * half;
        int n = nt * 16 + q15;
        red[wid][m * HD_ + n] = acc[mt][nt][g];
      }
  __syncthreads();
  for (int idx = tid; idx < MD_ * HD_; idx += 128) {
    float v = red[0][idx] + red[1][idx] + red[2][idx] + red[3][idx];
    sbf[(size_t)(b * H_ + h) * (MD_ * HD_) + idx] = (__bf16)v;
  }
}

// ---------------------------------------------------------------------------
// o = q @ s; RMSNorm over head dim; * g_norm_w; * silu(g)  -> X bf16
// grid (LQ/128, H, B); 8 waves, each one 16-row tile x 64 cols (4 WMMA, K=32).
// ---------------------------------------------------------------------------
__global__ __launch_bounds__(256)
void out_head_kernel(const __bf16* __restrict__ Qp, const __bf16* __restrict__ sbf,
                     const float* __restrict__ Gp, const float* __restrict__ gnw,
                     __bf16* __restrict__ X) {
  const int mblk = blockIdx.x, h = blockIdx.y, b = blockIdx.z;
  const int tid  = threadIdx.x;
  const int wid  = tid >> 5;
  const int lane = tid & 31;
  const int half = lane >> 4;
  const int q15  = lane & 15;
  const int mbase = mblk * 128 + wid * 16;

  const size_t sB = (size_t)(b * H_ + h) * (MD_ * HD_);
  FragBF bf[4];
  for (int nt = 0; nt < 4; ++nt) {              // B = s: [32 k][64 n]
    int c = nt * 16 + q15;
    for (int i = 0; i < 8; ++i) {
      int k = pairK(i, half);
      bf[nt].h[2 * i]     = *(const unsigned short*)(sbf + sB + (size_t)k * HD_ + c);
      bf[nt].h[2 * i + 1] = *(const unsigned short*)(sbf + sB + (size_t)(k + 1) * HD_ + c);
    }
  }
  FragBF af;                                    // A = q: row q15, K contiguous
  const size_t qRow = (size_t)(b * LQ_ + mbase + q15) * QK_ + h * MD_;
  for (int i = 0; i < 8; ++i)
    af.u[i] = *(const unsigned int*)(Qp + qRow + pairK(i, half));

  v8f acc[4];
  for (int nt = 0; nt < 4; ++nt) acc[nt] = (v8f)(0.0f);
  for (int nt = 0; nt < 4; ++nt)
    acc[nt] = __builtin_amdgcn_wmma_f32_16x16x32_bf16(
        false, af.v, false, bf[nt].v, (short)0, acc[nt], false, false);

  // RMSNorm: each lane holds rows {g+8*half} for column q15 of 4 n-tiles.
  float ss[8];
  for (int g = 0; g < 8; ++g) {
    float s = 0.f;
    for (int nt = 0; nt < 4; ++nt) { float v = acc[nt][g]; s += v * v; }
    // Reduce across the 16 lanes of this half-group (masks < 16 stay inside).
    for (int mask = 1; mask < 16; mask <<= 1) s += __shfl_xor(s, mask, 32);
    ss[g] = s;
  }
  for (int g = 0; g < 8; ++g) {
    float scale = rsqrtf(ss[g] * (1.0f / HD_) + RMSEPS_);
    int m = mbase + g + 8 * half;
    for (int nt = 0; nt < 4; ++nt) {
      int col = nt * 16 + q15;
      float v = acc[nt][g] * scale * gnw[col];
      float gt = Gp[(size_t)(b * LQ_ + m) * D_ + h * HD_ + col];
      v *= gt / (1.0f + expf(-gt));            // silu(g)
      X[(size_t)(b * LQ_ + m) * D_ + h * HD_ + col] = (__bf16)v;
    }
  }
}

// ---------------------------------------------------------------------------
// Host launcher
// ---------------------------------------------------------------------------
extern "C" void kernel_launch(void* const* d_in, const int* in_sizes, int n_in,
                              void* d_out, int out_size, void* d_ws, size_t ws_size,
                              hipStream_t stream) {
  const float* query = (const float*)d_in[0];
  const float* key   = (const float*)d_in[1];
  const float* value = (const float*)d_in[2];
  const float* Wq    = (const float*)d_in[3];
  const float* Wk    = (const float*)d_in[4];
  const float* Wv    = (const float*)d_in[5];
  const float* Wg    = (const float*)d_in[6];
  const float* Wgk1  = (const float*)d_in[7];
  const float* Wgk2  = (const float*)d_in[8];
  const float* bgk   = (const float*)d_in[9];
  const float* Wo    = (const float*)d_in[10];
  const float* gnw   = (const float*)d_in[11];
  float* out = (float*)d_out;

  char* w = (char*)d_ws;
  size_t off = 0;
  auto alloc = [&](size_t bytes) {
    void* p = w + off;
    off = (off + bytes + 255) & ~(size_t)255;
    return p;
  };

  const size_t nAct = (size_t)ROWS_ * D_;          // 8,388,608
  __bf16* qb   = (__bf16*)alloc(nAct * 2);
  __bf16* kb   = (__bf16*)alloc(nAct * 2);
  __bf16* vb   = (__bf16*)alloc(nAct * 2);
  __bf16* Wq_b = (__bf16*)alloc((size_t)D_ * QK_ * 2);
  __bf16* Wk_b = (__bf16*)alloc((size_t)D_ * QK_ * 2);
  __bf16* Wv_b = (__bf16*)alloc((size_t)D_ * D_ * 2);
  __bf16* Wg_b = (__bf16*)alloc((size_t)D_ * D_ * 2);
  __bf16* Wo_b = (__bf16*)alloc((size_t)D_ * D_ * 2);
  __bf16* Qp   = (__bf16*)alloc((size_t)ROWS_ * QK_ * 2);
  __bf16* Kp   = (__bf16*)alloc((size_t)ROWS_ * QK_ * 2);
  __bf16* Vp   = (__bf16*)alloc((size_t)ROWS_ * D_ * 2);
  float*  Gp   = (float*)alloc((size_t)ROWS_ * D_ * 4);
  float*  gkb  = (float*)alloc((size_t)ROWS_ * QK_ * 4);
  float*  cs   = (float*)alloc((size_t)B_ * NCH * QK_ * 4);
  float*  co   = (float*)alloc((size_t)B_ * NCH * QK_ * 4);
  __bf16* kd   = (__bf16*)alloc((size_t)ROWS_ * QK_ * 2);
  __bf16* sbf  = (__bf16*)alloc((size_t)B_ * H_ * MD_ * HD_ * 2);
  __bf16* X    = (__bf16*)alloc(nAct * 2);
  (void)ws_size; (void)n_in; (void)in_sizes; (void)out_size;

  // --- convert activations + big weights to bf16 ---
  cvt_f32_bf16<<<4096, 256, 0, stream>>>(query, qb, (int)nAct);
  cvt_f32_bf16<<<4096, 256, 0, stream>>>(key,   kb, (int)nAct);
  cvt_f32_bf16<<<4096, 256, 0, stream>>>(value, vb, (int)nAct);
  cvt_f32_bf16<<<512,  256, 0, stream>>>(Wq, Wq_b, D_ * QK_);
  cvt_f32_bf16<<<512,  256, 0, stream>>>(Wk, Wk_b, D_ * QK_);
  cvt_f32_bf16<<<1024, 256, 0, stream>>>(Wv, Wv_b, D_ * D_);
  cvt_f32_bf16<<<1024, 256, 0, stream>>>(Wg, Wg_b, D_ * D_);
  cvt_f32_bf16<<<1024, 256, 0, stream>>>(Wo, Wo_b, D_ * D_);

  // --- projections (WMMA GEMMs, TDM-fed A tiles) ---
  gemm_bf16_kernel<<<dim3(QK_ / BN, ROWS_ / BM), 256, 0, stream>>>(
      qb, Wq_b, Qp, ROWS_, QK_, D_, 1);
  gemm_bf16_kernel<<<dim3(QK_ / BN, ROWS_ / BM), 256, 0, stream>>>(
      kb, Wk_b, Kp, ROWS_, QK_, D_, 1);
  gemm_bf16_kernel<<<dim3(D_ / BN, ROWS_ / BM), 256, 0, stream>>>(
      vb, Wv_b, Vp, ROWS_, D_, D_, 1);
  gemm_bf16_kernel<<<dim3(D_ / BN, ROWS_ / BM), 256, 0, stream>>>(
      qb, Wg_b, Gp, ROWS_, D_, D_, 0);

  // --- low-rank gate projection + logsigmoid ---
  gk_proj_kernel<<<ROWS_, 256, 0, stream>>>(value, Wgk1, Wgk2, bgk, gkb);

  // --- chunked scan: decay = exp(flip(cumsum(gk))), kd = k * decay ---
  scan_chunksum<<<dim3(NCH, B_), 512, 0, stream>>>(gkb, cs);
  scan_chunkoff<<<B_, 512, 0, stream>>>(cs, co);
  scan_apply<<<dim3(NCH, B_), 512, 0, stream>>>(gkb, co, Kp, kd);

  // --- per-head state and output ---
  state_kernel<<<dim3(H_, B_), 128, 0, stream>>>(kd, Vp, sbf);
  out_head_kernel<<<dim3(LQ_ / 128, H_, B_), 256, 0, stream>>>(
      Qp, sbf, Gp, gnw, X);

  // --- final projection into fp32 output ---
  gemm_bf16_kernel<<<dim3(D_ / BN, ROWS_ / BM), 256, 0, stream>>>(
      X, Wo_b, out, ROWS_, D_, D_, 0);
}
